// RevisedRWKV_DualBranch_55387898250025
// MI455X (gfx1250) — compile-verified, matrix operand order checked by reading, not run on hardware
//
#include <hip/hip_runtime.h>
#include <math.h>

// ---------------------------------------------------------------------------
// RevisedRWKV dual-branch forward for MI455X (gfx1250, wave32, WMMA).
// All GEMMs run as bf16 x bf16 -> f32 via v_wmma_f32_16x16x32_bf16, software
// pipelined so fragment loads overlap the XDL pipe.
// ---------------------------------------------------------------------------

#define HID    256
#define DFF    1024
#define BATCH  8
#define TLEN   4096
#define LYR    6
#define NFREQ  129
#define NFRAME 65
#define SFREQ  (NFREQ * NFRAME)        // 8385
#define NT_T   (BATCH * TLEN)          // 32768 tokens (time branch)
#define NT_F   (BATCH * SFREQ)         // 67080 tokens (freq branch)
#define NCH    64                      // scan parallel chunks per sequence

typedef __bf16 bf16;
typedef __attribute__((ext_vector_type(16))) __bf16 v16bf;
typedef __attribute__((ext_vector_type(8)))  float  v8f;

__device__ __forceinline__ float bf2f(bf16 x) { return (float)x; }
__device__ __forceinline__ bf16  f2bf(float x) { return (bf16)x; }
__device__ __forceinline__ float sigmoidf_(float x) { return 1.f / (1.f + __expf(-x)); }

// ---------------------------------------------------------------------------
// WMMA GEMM: C[M,N] = epilogue( A[M,K] @ W[N,K]^T )
// A, W are bf16 row-major (both contiguous over K, so both load identically).
// Per-wave tile: 32 rows x 64 cols -> 2 A fragments, 4 B fragments, 8 WMMAs
// per k-step (each B fragment reused twice). Block: 8 waves = 256 x 64 tile.
// Fragment layout per CDNA5 ISA 7.12.2 (16-bit A 16x32): lane m<16 holds row m,
// K = {0..7, 16..23}; lane m+16 holds row m, K = {8..15, 24..31}. B symmetric
// with N in place of M. Two 16B loads per fragment per k-step.
// Software pipelined: next k-step fragments are fetched into fresh registers
// before the current step's WMMAs so VMEM overlaps the matrix pipe (partial
// s_wait_loadcnt instead of 0).
// ACT: 0=none 1=relu 2=relu^2 3=sigmoid. Optional per-col scale/shift (BN/bias),
// residual (out = res + v) and gate (out = res + gate*v), bf16 or f32 store.
// ---------------------------------------------------------------------------
union Frag { v16bf v; uint4 u[2]; };

#define LOAD_FRAG(fr, ptr, kk)                                   \
    do {                                                         \
        (fr).u[0] = *(const uint4*)((ptr) + (kk));               \
        (fr).u[1] = *(const uint4*)((ptr) + (kk) + 16);          \
    } while (0)

#define WMMA_BF16(a_, b_, c_)                                    \
    __builtin_amdgcn_wmma_f32_16x16x32_bf16(false, (a_).v, false, (b_).v, (short)0, (c_), false, false)

template <int ACT, bool OUT_BF16, bool HAS_RES, bool HAS_GATE>
__global__ __launch_bounds__(256)
void k_gemm(const bf16* __restrict__ A, const bf16* __restrict__ W,
            float* __restrict__ Cf, bf16* __restrict__ Cb,
            const float* __restrict__ scale, const float* __restrict__ shift,
            const float* __restrict__ res, const float* __restrict__ gate,
            int M, int N, int K)
{
    const int lane = threadIdx.x & 31;
    const int wave = threadIdx.x >> 5;
    const int m0 = blockIdx.y * 256 + wave * 32;
    const int n0 = blockIdx.x * 64;
    const int mh = lane & 15;
    const int hi = lane >> 4;

    int rowA0 = m0 + mh;      if (rowA0 >= M) rowA0 = M - 1;
    int rowA1 = m0 + 16 + mh; if (rowA1 >= M) rowA1 = M - 1;
    const bf16* ap0 = A + (size_t)rowA0 * K + hi * 8;
    const bf16* ap1 = A + (size_t)rowA1 * K + hi * 8;
    int rb0 = n0 + mh;      if (rb0 >= N) rb0 = N - 1;
    int rb1 = n0 + 16 + mh; if (rb1 >= N) rb1 = N - 1;
    int rb2 = n0 + 32 + mh; if (rb2 >= N) rb2 = N - 1;
    int rb3 = n0 + 48 + mh; if (rb3 >= N) rb3 = N - 1;
    const bf16* wp0 = W + (size_t)rb0 * K + hi * 8;
    const bf16* wp1 = W + (size_t)rb1 * K + hi * 8;
    const bf16* wp2 = W + (size_t)rb2 * K + hi * 8;
    const bf16* wp3 = W + (size_t)rb3 * K + hi * 8;

    v8f acc[8];
#pragma unroll
    for (int i = 0; i < 8; ++i)
#pragma unroll
        for (int j = 0; j < 8; ++j) acc[i][j] = 0.f;

    // pipeline prologue: fragments for k = 0
    Frag a0, a1, b0, b1, b2, b3;
    LOAD_FRAG(a0, ap0, 0); LOAD_FRAG(a1, ap1, 0);
    LOAD_FRAG(b0, wp0, 0); LOAD_FRAG(b1, wp1, 0);
    LOAD_FRAG(b2, wp2, 0); LOAD_FRAG(b3, wp3, 0);

    for (int k = 32; k < K; k += 32) {
        Frag a0n, a1n, b0n, b1n, b2n, b3n;
        LOAD_FRAG(a0n, ap0, k); LOAD_FRAG(a1n, ap1, k);
        LOAD_FRAG(b0n, wp0, k); LOAD_FRAG(b1n, wp1, k);
        LOAD_FRAG(b2n, wp2, k); LOAD_FRAG(b3n, wp3, k);
        if (k + 64 < K) {                       // hint the streamed A rows
            __builtin_prefetch(ap0 + k + 64, 0, 1);
            __builtin_prefetch(ap1 + k + 64, 0, 1);
        }
        acc[0] = WMMA_BF16(a0, b0, acc[0]);
        acc[1] = WMMA_BF16(a0, b1, acc[1]);
        acc[2] = WMMA_BF16(a0, b2, acc[2]);
        acc[3] = WMMA_BF16(a0, b3, acc[3]);
        acc[4] = WMMA_BF16(a1, b0, acc[4]);
        acc[5] = WMMA_BF16(a1, b1, acc[5]);
        acc[6] = WMMA_BF16(a1, b2, acc[6]);
        acc[7] = WMMA_BF16(a1, b3, acc[7]);
        a0 = a0n; a1 = a1n; b0 = b0n; b1 = b1n; b2 = b2n; b3 = b3n;
    }
    // pipeline epilogue
    acc[0] = WMMA_BF16(a0, b0, acc[0]);
    acc[1] = WMMA_BF16(a0, b1, acc[1]);
    acc[2] = WMMA_BF16(a0, b2, acc[2]);
    acc[3] = WMMA_BF16(a0, b3, acc[3]);
    acc[4] = WMMA_BF16(a1, b0, acc[4]);
    acc[5] = WMMA_BF16(a1, b1, acc[5]);
    acc[6] = WMMA_BF16(a1, b2, acc[6]);
    acc[7] = WMMA_BF16(a1, b3, acc[7]);

#pragma unroll
    for (int nt = 0; nt < 4; ++nt) {
        int col = n0 + nt * 16 + mh;
        if (col >= N) continue;
        float scv = scale ? scale[col] : 1.f;
        float shv = shift ? shift[col] : 0.f;
#pragma unroll
        for (int ms = 0; ms < 2; ++ms) {
#pragma unroll
            for (int r = 0; r < 8; ++r) {
                int row = m0 + ms * 16 + hi * 8 + r;
                if (row >= M) continue;
                float v = acc[ms * 4 + nt][r] * scv + shv;
                if (ACT == 1) v = fmaxf(v, 0.f);
                if (ACT == 2) { v = fmaxf(v, 0.f); v = v * v; }
                if (ACT == 3) v = sigmoidf_(v);
                size_t idx = (size_t)row * N + col;
                if (HAS_GATE)      v = res[idx] + gate[idx] * v;
                else if (HAS_RES)  v = res[idx] + v;
                if (OUT_BF16) Cb[idx] = f2bf(v);
                else          Cf[idx] = v;
            }
        }
    }
}

static void launch_gemm(hipStream_t s, const bf16* A, const bf16* W, void* C, bool outbf,
                        const float* scale, const float* shift,
                        const float* res, const float* gate,
                        int M, int N, int K, int act)
{
    dim3 grid((N + 63) / 64, (M + 255) / 256), blk(256);
    float* Cf = (float*)C; bf16* Cb = (bf16*)C;
    if (gate)
        k_gemm<0, false, true, true><<<grid, blk, 0, s>>>(A, W, Cf, Cb, scale, shift, res, gate, M, N, K);
    else if (res)
        k_gemm<0, false, true, false><<<grid, blk, 0, s>>>(A, W, Cf, Cb, scale, shift, res, nullptr, M, N, K);
    else if (outbf) {
        if (act == 1)      k_gemm<1, true, false, false><<<grid, blk, 0, s>>>(A, W, Cf, Cb, scale, shift, nullptr, nullptr, M, N, K);
        else if (act == 2) k_gemm<2, true, false, false><<<grid, blk, 0, s>>>(A, W, Cf, Cb, scale, shift, nullptr, nullptr, M, N, K);
        else               k_gemm<0, true, false, false><<<grid, blk, 0, s>>>(A, W, Cf, Cb, scale, shift, nullptr, nullptr, M, N, K);
    } else {
        if (act == 1)      k_gemm<1, false, false, false><<<grid, blk, 0, s>>>(A, W, Cf, Cb, scale, shift, nullptr, nullptr, M, N, K);
        else if (act == 3) k_gemm<3, false, false, false><<<grid, blk, 0, s>>>(A, W, Cf, Cb, scale, shift, nullptr, nullptr, M, N, K);
        else               k_gemm<0, false, false, false><<<grid, blk, 0, s>>>(A, W, Cf, Cb, scale, shift, nullptr, nullptr, M, N, K);
    }
}

// ---------------------------------------------------------------------------
// Elementwise / reduction kernels
// ---------------------------------------------------------------------------
__global__ void k_convert(const float* __restrict__ src, bf16* __restrict__ dst, int n)
{
    int i = blockIdx.x * 256 + threadIdx.x;
    if (i < n) dst[i] = f2bf(src[i]);
}

// Fold BatchNorm(eval) + conv/linear bias into per-channel scale/shift.
__global__ void k_bnprep(const float* g, const float* be, const float* m, const float* v,
                         const float* bias, float* sc, float* sh, int n)
{
    int i = blockIdx.x * 256 + threadIdx.x;
    if (i >= n) return;
    float s = g[i] * rsqrtf(v[i] + 1e-5f);
    float b0 = bias ? bias[i] : 0.f;
    sc[i] = s;
    sh[i] = be[i] + (b0 - m[i]) * s;
}

// LayerNorm over 256 channels, one wave per token. g/b optional (null -> 1/0).
__global__ void k_ln(const float* __restrict__ in, const float* __restrict__ g,
                     const float* __restrict__ b, float* __restrict__ outF,
                     bf16* __restrict__ outB, int NT)
{
    int lane = threadIdx.x & 31, wave = threadIdx.x >> 5;
    int row = blockIdx.x * 8 + wave;
    if (row >= NT) return;
    const float* p = in + (size_t)row * HID;
    float x[8], s = 0.f, s2 = 0.f;
#pragma unroll
    for (int i = 0; i < 8; ++i) { x[i] = p[lane + 32 * i]; s += x[i]; s2 += x[i] * x[i]; }
#pragma unroll
    for (int o = 16; o > 0; o >>= 1) { s += __shfl_xor(s, o, 32); s2 += __shfl_xor(s2, o, 32); }
    float mean = s * (1.f / HID);
    float var = s2 * (1.f / HID) - mean * mean;
    float rstd = rsqrtf(var + 1e-5f);
#pragma unroll
    for (int i = 0; i < 8; ++i) {
        int c = lane + 32 * i;
        float y = (x[i] - mean) * rstd;
        if (g) y = y * g[c] + b[c];
        size_t idx = (size_t)row * HID + c;
        if (outF) outF[idx] = y;
        if (outB) outB[idx] = f2bf(y);
    }
}

// Two back-to-back affine LayerNorms (post-attention fuse), bf16 out.
__global__ void k_ln2x(const float* __restrict__ in,
                       const float* g1, const float* b1,
                       const float* g2, const float* b2,
                       bf16* __restrict__ outB, int NT)
{
    int lane = threadIdx.x & 31, wave = threadIdx.x >> 5;
    int row = blockIdx.x * 8 + wave;
    if (row >= NT) return;
    const float* p = in + (size_t)row * HID;
    float x[8], s = 0.f, s2 = 0.f;
#pragma unroll
    for (int i = 0; i < 8; ++i) { x[i] = p[lane + 32 * i]; s += x[i]; s2 += x[i] * x[i]; }
#pragma unroll
    for (int o = 16; o > 0; o >>= 1) { s += __shfl_xor(s, o, 32); s2 += __shfl_xor(s2, o, 32); }
    float mean = s * (1.f / HID);
    float rstd = rsqrtf(s2 * (1.f / HID) - mean * mean + 1e-5f);
    float y[8], t = 0.f, t2 = 0.f;
#pragma unroll
    for (int i = 0; i < 8; ++i) {
        int c = lane + 32 * i;
        y[i] = (x[i] - mean) * rstd * g1[c] + b1[c];
        t += y[i]; t2 += y[i] * y[i];
    }
#pragma unroll
    for (int o = 16; o > 0; o >>= 1) { t += __shfl_xor(t, o, 32); t2 += __shfl_xor(t2, o, 32); }
    float mean2 = t * (1.f / HID);
    float rstd2 = rsqrtf(t2 * (1.f / HID) - mean2 * mean2 + 1e-5f);
#pragma unroll
    for (int i = 0; i < 8; ++i) {
        int c = lane + 32 * i;
        outB[(size_t)row * HID + c] = f2bf((y[i] - mean2) * rstd2 * g2[c] + b2[c]);
    }
}

// Token-shift mix: out = ln[t]*tm + prev*(1-tm); prev = state (t==0) else ln[t-1].
// Emits both v-mix and r-mix as bf16. One block per token row.
__global__ void k_mix(const float* __restrict__ ln, const float* __restrict__ state,
                      const float* __restrict__ ta, const float* __restrict__ tb,
                      bf16* __restrict__ outA, bf16* __restrict__ outB, int S)
{
    int row = blockIdx.x, c = threadIdx.x;
    int t = row % S, b = row / S;
    float cur = ln[(size_t)row * HID + c];
    float prev;
    if (t == 0) prev = state ? state[b * HID + c] : 0.f;
    else        prev = ln[(size_t)(row - 1) * HID + c];
    float a0 = ta[c], b0 = tb[c];
    size_t idx = (size_t)row * HID + c;
    outA[idx] = f2bf(cur * a0 + prev * (1.f - a0));
    outB[idx] = f2bf(cur * b0 + prev * (1.f - b0));
}

__global__ void k_getstate(const float* __restrict__ x, float* __restrict__ st, int S)
{
    int i = blockIdx.x * 256 + threadIdx.x;          // BATCH*HID threads
    int b = i >> 8, c = i & 255;
    st[i] = x[((size_t)b * S + (S - 1)) * HID + c];
}

// states = sigmoid(r)*states + 0.1*v   (sigmoid already applied by GEMM epilogue)
__global__ void k_wkvgate(float* __restrict__ st, const float* __restrict__ rs,
                          const float* __restrict__ vb)
{
    size_t i = (size_t)blockIdx.x * 256 + threadIdx.x;
    st[i] = rs[i] * st[i] + 0.1f * vb[i];
}

// ---- 3-pass chunked exponential-decay scan: s_t = d*s_{t-1} + (1-d)*v_t ----
__global__ void k_scan_local(const float* __restrict__ v, float* __restrict__ st,
                             float* __restrict__ carry, const float* __restrict__ td,
                             int S, int CS)
{
    int gid = blockIdx.x * 256 + threadIdx.x;        // BATCH*NCH*HID
    int c = gid & 255, ch = (gid >> 8) & (NCH - 1), b = gid >> 14;
    float d = __expf(td[c]), om = 1.f - d, s = 0.f;
    int t0 = ch * CS, t1 = t0 + CS; if (t1 > S) t1 = S;
    for (int t = t0; t < t1; ++t) {
        size_t idx = ((size_t)b * S + t) * HID + c;
        s = d * s + om * v[idx];
        st[idx] = s;
    }
    carry[gid] = s;
}

__global__ void k_scan_carry(const float* __restrict__ carry, float* __restrict__ inits,
                             const float* __restrict__ td, int S, int CS)
{
    int gid = blockIdx.x * 256 + threadIdx.x;        // BATCH*HID
    int c = gid & 255, b = gid >> 8;
    float d = __expf(td[c]), init = 0.f;
    for (int ch = 0; ch < NCH; ++ch) {
        int base = (b * NCH + ch) * HID + c;
        inits[base] = init;
        int len = S - ch * CS; if (len > CS) len = CS; if (len < 0) len = 0;
        init = (len > 0 ? carry[base] : 0.f) + init * __powf(d, (float)len);
    }
}

__global__ void k_scan_fix(float* __restrict__ st, const float* __restrict__ inits,
                           const float* __restrict__ td, int S, int CS)
{
    int gid = blockIdx.x * 256 + threadIdx.x;
    int c = gid & 255, ch = (gid >> 8) & (NCH - 1), b = gid >> 14;
    float init = inits[gid];
    if (init == 0.f) return;
    float d = __expf(td[c]), p = init;
    int t0 = ch * CS, t1 = t0 + CS; if (t1 > S) t1 = S;
    for (int t = t0; t < t1; ++t) {
        p *= d;
        st[((size_t)b * S + t) * HID + c] += p;
    }
}

// ---- Time-branch front end: conv1d + folded BN + relu ----
__global__ void k_conv1(const float* __restrict__ x, const float* __restrict__ w,
                        const float* __restrict__ sc, const float* __restrict__ sh,
                        float* __restrict__ h1)
{
    int gid = blockIdx.x * 256 + threadIdx.x;        // 8*128*4096
    int t = gid & 4095, c = (gid >> 12) & 127, b = gid >> 19;
    float acc = 0.f;
#pragma unroll
    for (int j = 0; j < 7; ++j) {
        int ti = t + j - 3;
        if (ti >= 0 && ti < TLEN) acc += w[c * 7 + j] * x[b * TLEN + ti];
    }
    h1[((size_t)(b * 128 + c) << 12) + t] = fmaxf(acc * sc[c] + sh[c], 0.f);
}

__global__ void k_conv2(const float* __restrict__ h1, const float* __restrict__ w,
                        const float* __restrict__ sc, const float* __restrict__ sh,
                        float* __restrict__ out)     // token-major [B*T, 256]
{
    int gid = blockIdx.x * 256 + threadIdx.x;        // 8*4096*256
    int c = gid & 255, t = (gid >> 8) & 4095, b = gid >> 20;
    float acc = 0.f;
    for (int ch = 0; ch < 128; ++ch) {
        const float* hp = h1 + ((size_t)(b * 128 + ch) << 12);
        const float* wp = w + (c * 128 + ch) * 5;
#pragma unroll
        for (int j = 0; j < 5; ++j) {
            int ti = t + j - 2;
            if (ti >= 0 && ti < TLEN) acc += wp[j] * hp[ti];
        }
    }
    out[((size_t)(b * TLEN + t)) * HID + c] = fmaxf(acc * sc[c] + sh[c], 0.f);
}

// ---- STFT (hann, reflect pad, onesided) as naive 256-pt DFT, phase mod 256 ----
__global__ void k_stft(const float* __restrict__ x, float* __restrict__ out)
{
    int gid = blockIdx.x * 256 + threadIdx.x;        // B*129*65
    if (gid >= NT_F) return;
    int fr = gid % NFRAME, f = (gid / NFRAME) % NFREQ, b = gid / (NFRAME * NFREQ);
    const float TWO_PI = 6.283185307179586f;
    float re = 0.f, im = 0.f;
    for (int n = 0; n < 256; ++n) {
        int idx = fr * 64 + n - 128;                 // reflect pad 128
        if (idx < 0) idx = -idx;
        if (idx > TLEN - 1) idx = 2 * (TLEN - 1) - idx;
        float win = 0.5f - 0.5f * __cosf(TWO_PI * n * (1.f / 256.f));
        float wx = x[b * TLEN + idx] * win;
        int ph = (f * n) & 255;
        float sn, cs;
        __sincosf(-TWO_PI * ph * (1.f / 256.f), &sn, &cs);
        re += wx * cs;
        im += wx * sn;
    }
    size_t row = (size_t)b * SFREQ + f * NFRAME + fr;  // token = f*Fr + fr (F-major)
    out[row * 2 + 0] = re;
    out[row * 2 + 1] = im;
}

// 2->128 channel projection + folded BN + relu, token-major output.
__global__ void k_proj1(const float* __restrict__ st, const float* __restrict__ w1,
                        const float* __restrict__ sc, const float* __restrict__ sh,
                        float* __restrict__ out)
{
    int gid = blockIdx.x * 256 + threadIdx.x;        // NT_F*128
    int c = gid & 127;
    size_t row = (size_t)(gid >> 7);
    float v = w1[c * 2 + 0] * st[row * 2 + 0] + w1[c * 2 + 1] * st[row * 2 + 1];
    out[row * 128 + c] = fmaxf(v * sc[c] + sh[c], 0.f);
}

// F.interpolate(mode='linear', align_corners=False): [B,Sf,C] -> [B,T,C]
__global__ void k_interp(const float* __restrict__ fx, float* __restrict__ out)
{
    int row = blockIdx.x, c = threadIdx.x;           // NT_T rows
    int b = row >> 12, t = row & 4095;
    float pos = (t + 0.5f) * ((float)SFREQ / (float)TLEN) - 0.5f;
    pos = fminf(fmaxf(pos, 0.f), (float)(SFREQ - 1));
    int i0 = (int)floorf(pos);
    int i1 = i0 + 1; if (i1 > SFREQ - 1) i1 = SFREQ - 1;
    float w = pos - (float)i0;
    float v0 = fx[((size_t)b * SFREQ + i0) * HID + c];
    float v1 = fx[((size_t)b * SFREQ + i1) * HID + c];
    out[(size_t)row * HID + c] = v0 * (1.f - w) + v1 * w;
}

// MHA core: seq axis = batch dim B (len 8), batch = T, 8 heads x 32 dims.
__global__ void k_attn(const float* __restrict__ q, const float* __restrict__ k,
                       const float* __restrict__ v, float* __restrict__ o, int T)
{
    int gid = blockIdx.x * 256 + threadIdx.x;        // T*8h*8i
    if (gid >= T * 64) return;
    int i = gid & 7, h = (gid >> 3) & 7, t = gid >> 6;
    size_t qoff = ((size_t)i * T + t) * HID + h * 32;
    float qv[32];
#pragma unroll
    for (int d = 0; d < 32; ++d) qv[d] = q[qoff + d];
    float sc[8], mx = -1e30f;
#pragma unroll
    for (int j = 0; j < 8; ++j) {
        size_t off = ((size_t)j * T + t) * HID + h * 32;
        float s = 0.f;
#pragma unroll
        for (int d = 0; d < 32; ++d) s += qv[d] * k[off + d];
        s *= 0.17677669529663687f;                   // 1/sqrt(32)
        sc[j] = s; mx = fmaxf(mx, s);
    }
    float den = 0.f;
#pragma unroll
    for (int j = 0; j < 8; ++j) { sc[j] = __expf(sc[j] - mx); den += sc[j]; }
    float inv = 1.f / den;
    float ov[32];
#pragma unroll
    for (int d = 0; d < 32; ++d) ov[d] = 0.f;
#pragma unroll
    for (int j = 0; j < 8; ++j) {
        float w = sc[j] * inv;
        size_t off = ((size_t)j * T + t) * HID + h * 32;
#pragma unroll
        for (int d = 0; d < 32; ++d) ov[d] += w * v[off + d];
    }
#pragma unroll
    for (int d = 0; d < 32; ++d) o[qoff + d] = ov[d];
}

// Head: out[row] = o2[row,:64] . w3 + b3
__global__ void k_head(const bf16* __restrict__ o2, const float* __restrict__ w3,
                       const float* __restrict__ b3, float* __restrict__ out)
{
    int row = blockIdx.x * 256 + threadIdx.x;        // NT_T
    float acc = b3[0];
#pragma unroll
    for (int i = 0; i < 64; ++i) acc += bf2f(o2[(size_t)row * 64 + i]) * w3[i];
    out[row] = acc;
}

// ---------------------------------------------------------------------------
// Host orchestration
// ---------------------------------------------------------------------------
struct RwkvCtx {
    hipStream_t s;
    float* x; int S; int NT;
    const float *tmv, *tmr, *cmk, *cmr, *ln1g, *ln1b, *ln2g, *ln2b, *td;
    const bf16 *Wv, *Wr, *Wo, *Wkc, *Wrc, *Wvc;
    float *lnb, *vb, *rsb, *stb, *statebuf, *carry, *inits;
    bf16 *mA, *mB, *wkvb, *kcb;
};

static void run_rwkv(const RwkvCtx& c)
{
    const int NT = c.NT;
    const int CS = (c.S + NCH - 1) / NCH;
    dim3 blk(256);
    for (int l = 0; l < LYR; ++l) {
        const float* st0 = nullptr;
        if (l > 0) {  // state for layer l = this layer's input x at last token
            k_getstate<<<(BATCH * HID) / 256, blk, 0, c.s>>>(c.x, c.statebuf, c.S);
            st0 = c.statebuf;
        }
        // --- token mixing ---
        k_ln<<<(NT + 7) / 8, blk, 0, c.s>>>(c.x, c.ln1g + l * HID, c.ln1b + l * HID,
                                            c.lnb, (bf16*)nullptr, NT);
        k_mix<<<NT, blk, 0, c.s>>>(c.lnb, st0, c.tmv + l * HID, c.tmr + l * HID,
                                   c.mA, c.mB, c.S);
        launch_gemm(c.s, c.mA, c.Wv + (size_t)l * HID * HID, c.vb, false,
                    nullptr, nullptr, nullptr, nullptr, NT, HID, HID, 0);
        launch_gemm(c.s, c.mB, c.Wr + (size_t)l * HID * HID, c.rsb, false,
                    nullptr, nullptr, nullptr, nullptr, NT, HID, HID, 3);   // sigmoid(r)
        // --- decay scan (3-pass, 64-way parallel per channel) ---
        k_scan_local<<<(BATCH * NCH * HID) / 256, blk, 0, c.s>>>(c.vb, c.stb, c.carry,
                                                                 c.td + l * HID, c.S, CS);
        k_scan_carry<<<(BATCH * HID) / 256, blk, 0, c.s>>>(c.carry, c.inits,
                                                           c.td + l * HID, c.S, CS);
        k_scan_fix<<<(BATCH * NCH * HID) / 256, blk, 0, c.s>>>(c.stb, c.inits,
                                                               c.td + l * HID, c.S, CS);
        // wkv = sigmoid(r)*states + 0.1*v ; LN(no affine) -> bf16
        k_wkvgate<<<NT, blk, 0, c.s>>>(c.stb, c.rsb, c.vb);
        k_ln<<<(NT + 7) / 8, blk, 0, c.s>>>(c.stb, nullptr, nullptr, nullptr, c.wkvb, NT);
        // x += wkv @ Wo^T  (residual epilogue, in-place)
        launch_gemm(c.s, c.wkvb, c.Wo + (size_t)l * HID * HID, c.x, false,
                    nullptr, nullptr, c.x, nullptr, NT, HID, HID, 0);
        // --- channel mixing ---
        k_ln<<<(NT + 7) / 8, blk, 0, c.s>>>(c.x, c.ln2g + l * HID, c.ln2b + l * HID,
                                            c.lnb, (bf16*)nullptr, NT);
        k_mix<<<NT, blk, 0, c.s>>>(c.lnb, st0, c.cmk + l * HID, c.cmr + l * HID,
                                   c.mA, c.mB, c.S);
        launch_gemm(c.s, c.mA, c.Wkc + (size_t)l * DFF * HID, c.kcb, true,
                    nullptr, nullptr, nullptr, nullptr, NT, DFF, HID, 2);   // relu^2 -> bf16
        launch_gemm(c.s, c.mB, c.Wrc + (size_t)l * HID * HID, c.rsb, false,
                    nullptr, nullptr, nullptr, nullptr, NT, HID, HID, 3);   // sigmoid(rc)
        // x += sigmoid(rc) * (relu(kc)^2 @ Wvc^T)
        launch_gemm(c.s, c.kcb, c.Wvc + (size_t)l * HID * DFF, c.x, false,
                    nullptr, nullptr, c.x, c.rsb, NT, HID, DFF, 0);
    }
}

extern "C" void kernel_launch(void* const* d_in, const int* in_sizes, int n_in,
                              void* d_out, int out_size, void* d_ws, size_t ws_size,
                              hipStream_t stream)
{
    (void)in_sizes; (void)n_in; (void)out_size; (void)ws_size;
    auto IN = [&](int i) { return (const float*)d_in[i]; };
    // Input order (recursive dict order of setup_inputs):
    // 0:x  1-12:tp(w1,b1,g1,be1,m1,v1,w2,b2,g2,be2,m2,v2)
    // 13-24:fp(w1,b1,g1,be1,m1,v1,w2,b2,g2,be2,m2,v2)
    // 25-41:tb(tmk,tmv,tmr,Wk,Wv,Wr,Wo,cmk,cmr,Wkc,Wrc,Wvc,ln1g,ln1b,ln2g,ln2b,td)
    // 42-58:fb(same)  59-66:attn(Wq,Wk,Wv,Wo,bq,bk,bv,bo)
    // 67-70:fuse(fng,fnb,log,lob)  71-76:op(w1,b1,w2,b2,w3,b3)

    char* base = (char*)d_ws;
    size_t off = 0;
    auto alloc = [&](size_t bytes) -> void* {
        off = (off + 255) & ~(size_t)255;
        void* p = base + off; off += bytes; return p;
    };

    // bf16 weight pools
    bf16 *WvT = (bf16*)alloc(6 * 65536 * 2),  *WrT = (bf16*)alloc(6 * 65536 * 2);
    bf16 *WoT = (bf16*)alloc(6 * 65536 * 2),  *WkcT = (bf16*)alloc(6 * 262144 * 2);
    bf16 *WrcT = (bf16*)alloc(6 * 65536 * 2), *WvcT = (bf16*)alloc(6 * 262144 * 2);
    bf16 *WvF = (bf16*)alloc(6 * 65536 * 2),  *WrF = (bf16*)alloc(6 * 65536 * 2);
    bf16 *WoF = (bf16*)alloc(6 * 65536 * 2),  *WkcF = (bf16*)alloc(6 * 262144 * 2);
    bf16 *WrcF = (bf16*)alloc(6 * 65536 * 2), *WvcF = (bf16*)alloc(6 * 262144 * 2);
    bf16 *Waq = (bf16*)alloc(65536 * 2), *Wak = (bf16*)alloc(65536 * 2);
    bf16 *Wav = (bf16*)alloc(65536 * 2), *Wao = (bf16*)alloc(65536 * 2);
    bf16 *Wop1 = (bf16*)alloc(32768 * 2), *Wop2 = (bf16*)alloc(8192 * 2);
    bf16 *Wfp2 = (bf16*)alloc(32768 * 2);
    // BN-folded constants
    float *c1s = (float*)alloc(128 * 4), *c1t = (float*)alloc(128 * 4);
    float *c2s = (float*)alloc(256 * 4), *c2t = (float*)alloc(256 * 4);
    float *f1s = (float*)alloc(128 * 4), *f1t = (float*)alloc(128 * 4);
    float *f2s = (float*)alloc(256 * 4), *f2t = (float*)alloc(256 * 4);
    // activations
    float *h1  = (float*)alloc((size_t)BATCH * 128 * TLEN * 4);
    float *xT  = (float*)alloc((size_t)NT_T * HID * 4);       // time branch x (persists)
    float *fxi = (float*)alloc((size_t)NT_T * HID * 4);       // interpolated freq_x
    float *xF  = (float*)alloc((size_t)NT_F * HID * 4);       // freq branch x / fused_pre
    float *lnb = (float*)alloc((size_t)NT_F * HID * 4);
    float *vb  = (float*)alloc((size_t)NT_F * HID * 4);
    float *rsb = (float*)alloc((size_t)NT_F * HID * 4);
    float *stb = (float*)alloc((size_t)NT_F * HID * 4);
    bf16  *mA  = (bf16*)alloc((size_t)NT_F * HID * 2);
    bf16  *mB  = (bf16*)alloc((size_t)NT_F * HID * 2);
    bf16  *wkvb = (bf16*)alloc((size_t)NT_F * HID * 2);
    bf16  *kcb = (bf16*)alloc((size_t)NT_F * DFF * 2);
    float *carry = (float*)alloc((size_t)BATCH * NCH * HID * 4);
    float *inits = (float*)alloc((size_t)BATCH * NCH * HID * 4);
    float *statebuf = (float*)alloc(BATCH * HID * 4);
    float *stftb = (float*)alloc((size_t)NT_F * 2 * 4);

    dim3 blk(256);
    auto cvt = [&](const float* src, bf16* dst, int n) {
        k_convert<<<(n + 255) / 256, blk, 0, stream>>>(src, dst, n);
    };

    // ---- weight conversion + BN folding ----
    cvt(IN(29), WvT, 6 * 65536);  cvt(IN(30), WrT, 6 * 65536);
    cvt(IN(31), WoT, 6 * 65536);  cvt(IN(34), WkcT, 6 * 262144);
    cvt(IN(35), WrcT, 6 * 65536); cvt(IN(36), WvcT, 6 * 262144);
    cvt(IN(46), WvF, 6 * 65536);  cvt(IN(47), WrF, 6 * 65536);
    cvt(IN(48), WoF, 6 * 65536);  cvt(IN(51), WkcF, 6 * 262144);
    cvt(IN(52), WrcF, 6 * 65536); cvt(IN(53), WvcF, 6 * 262144);
    cvt(IN(59), Waq, 65536); cvt(IN(60), Wak, 65536);
    cvt(IN(61), Wav, 65536); cvt(IN(62), Wao, 65536);
    cvt(IN(71), Wop1, 32768); cvt(IN(73), Wop2, 8192);
    cvt(IN(19), Wfp2, 32768);
    k_bnprep<<<1, blk, 0, stream>>>(IN(3), IN(4), IN(5), IN(6), IN(2), c1s, c1t, 128);
    k_bnprep<<<1, blk, 0, stream>>>(IN(9), IN(10), IN(11), IN(12), IN(8), c2s, c2t, 256);
    k_bnprep<<<1, blk, 0, stream>>>(IN(15), IN(16), IN(17), IN(18), IN(14), f1s, f1t, 128);
    k_bnprep<<<1, blk, 0, stream>>>(IN(21), IN(22), IN(23), IN(24), IN(20), f2s, f2t, 256);

    // ---- time branch: conv front end -> RWKV ----
    k_conv1<<<(BATCH * 128 * TLEN) / 256, blk, 0, stream>>>(IN(0), IN(1), c1s, c1t, h1);
    k_conv2<<<(BATCH * TLEN * HID) / 256, blk, 0, stream>>>(h1, IN(7), c2s, c2t, xT);
    RwkvCtx tc;
    tc.s = stream; tc.x = xT; tc.S = TLEN; tc.NT = NT_T;
    tc.tmv = IN(26); tc.tmr = IN(27); tc.cmk = IN(32); tc.cmr = IN(33);
    tc.ln1g = IN(37); tc.ln1b = IN(38); tc.ln2g = IN(39); tc.ln2b = IN(40); tc.td = IN(41);
    tc.Wv = WvT; tc.Wr = WrT; tc.Wo = WoT; tc.Wkc = WkcT; tc.Wrc = WrcT; tc.Wvc = WvcT;
    tc.lnb = lnb; tc.vb = vb; tc.rsb = rsb; tc.stb = stb; tc.statebuf = statebuf;
    tc.carry = carry; tc.inits = inits; tc.mA = mA; tc.mB = mB; tc.wkvb = wkvb; tc.kcb = kcb;
    run_rwkv(tc);

    // ---- freq branch: STFT -> 2->128->256 projections -> RWKV ----
    k_stft<<<(NT_F + 255) / 256, blk, 0, stream>>>(IN(0), stftb);
    k_proj1<<<(NT_F * 128) / 256, blk, 0, stream>>>(stftb, IN(13), f1s, f1t, vb);
    cvt(vb, mA, NT_F * 128);
    launch_gemm(stream, mA, Wfp2, xF, false, f2s, f2t, nullptr, nullptr, NT_F, HID, 128, 1);
    RwkvCtx fc = tc;
    fc.x = xF; fc.S = SFREQ; fc.NT = NT_F;
    fc.tmv = IN(43); fc.tmr = IN(44); fc.cmk = IN(49); fc.cmr = IN(50);
    fc.ln1g = IN(54); fc.ln1b = IN(55); fc.ln2g = IN(56); fc.ln2b = IN(57); fc.td = IN(58);
    fc.Wv = WvF; fc.Wr = WrF; fc.Wo = WoF; fc.Wkc = WkcF; fc.Wrc = WrcF; fc.Wvc = WvcF;
    run_rwkv(fc);

    // ---- interpolate freq_x to T, cross attention fuse ----
    k_interp<<<NT_T, blk, 0, stream>>>(xF, fxi);
    cvt(xT, mA, NT_T * HID);                          // time_x bf16
    cvt(fxi, mB, NT_T * HID);                         // kv bf16
    launch_gemm(stream, mA, Waq, lnb, false, nullptr, IN(63), nullptr, nullptr, NT_T, HID, HID, 0);
    launch_gemm(stream, mB, Wak, vb,  false, nullptr, IN(64), nullptr, nullptr, NT_T, HID, HID, 0);
    launch_gemm(stream, mB, Wav, rsb, false, nullptr, IN(65), nullptr, nullptr, NT_T, HID, HID, 0);
    k_attn<<<(TLEN * 64) / 256, blk, 0, stream>>>(lnb, vb, rsb, stb, TLEN);
    cvt(stb, wkvb, NT_T * HID);
    // fused_pre = attn_out @ Wo^T + bo + time_x  (residual epilogue), reuse xF
    launch_gemm(stream, wkvb, Wao, xF, false, nullptr, IN(66), xT, nullptr, NT_T, HID, HID, 0);
    bf16* fusedb = kcb;                               // carve bf16 scratch out of kcb pool
    bf16* o1b = fusedb + (size_t)NT_T * HID;
    bf16* o2b = o1b + (size_t)NT_T * 128;
    k_ln2x<<<(NT_T + 7) / 8, blk, 0, stream>>>(xF, IN(67), IN(68), IN(69), IN(70), fusedb, NT_T);

    // ---- output MLP 256->128->64->1 ----
    launch_gemm(stream, fusedb, Wop1, o1b, true, nullptr, IN(72), nullptr, nullptr, NT_T, 128, HID, 1);
    launch_gemm(stream, o1b, Wop2, o2b, true, nullptr, IN(74), nullptr, nullptr, NT_T, 64, 128, 1);
    k_head<<<NT_T / 256, blk, 0, stream>>>(o2b, IN(75), IN(76), (float*)d_out);
}
